// VimBackbone_18837726560383
// MI455X (gfx1250) — matrix operand order repaired and checked by reading, not compiled, not run on hardware
//
#include <hip/hip_runtime.h>

typedef __attribute__((ext_vector_type(16))) _Float16 v16h;
typedef __attribute__((ext_vector_type(8)))  _Float16 v8h;
typedef __attribute__((ext_vector_type(8)))  float    v8f;
typedef _Float16 half_t;

#define N_TOK 4096
#define DD    384
#define DDI   768
#define TWODI 1536
#define HMID  1536
#define NLAY  12
#define LDSS  40   // LDS row stride in halves (80B, 16B-aligned, bank-rotating)

// ---------------------------------------------------------------------------
// Transposing f32 -> f16 convert: in[K,N] (row-major) -> out[N,K] (row-major)
// Coalesced writes; strided reads stay in L2 (weights read once per layer).
// ---------------------------------------------------------------------------
__global__ __launch_bounds__(256) void cvt_tr(const float* __restrict__ in,
                                              half_t* __restrict__ out,
                                              int K, int N) {
  int total = K * N;
  for (int i = blockIdx.x * 256 + threadIdx.x; i < total; i += gridDim.x * 256) {
    int n = i / K, k = i - n * K;
    out[i] = (half_t)in[(size_t)k * N + n];
  }
}

// ---------------------------------------------------------------------------
// Patchify: x[B,3,256,256] -> patches_h[t=4096][cpp=768] (f16)
// ---------------------------------------------------------------------------
__global__ __launch_bounds__(256) void patchify(const float* __restrict__ x,
                                                half_t* __restrict__ out) {
  int t = blockIdx.x, tid = threadIdx.x;
  int bimg = t >> 8, pix = t & 255;
  int r = pix >> 4, c = pix & 15;
#pragma unroll
  for (int e = 0; e < 3; ++e) {
    int cpp = tid + 256 * e;              // < 768
    int ch = cpp >> 8, rem = cpp & 255;
    int pr = rem >> 4, pc = rem & 15;
    size_t xi = (((size_t)(bimg * 3 + ch)) * 256 + (r * 16 + pr)) * 256 + (c * 16 + pc);
    out[(size_t)t * 768 + cpp] = (half_t)x[xi];
  }
}

// ---------------------------------------------------------------------------
// WMMA GEMM: C[M,N] = A[M,K](f16) * Wt[N,K](f16, pre-transposed) + bias.
// Block = 256 threads (8 waves), tile 64x64, each wave 16 rows x 32 cols.
// Tiles staged into LDS with GLOBAL_LOAD_ASYNC_TO_LDS_B128 (ASYNCcnt).
// EPI: 0 = f32 store, 1 = f16 store, 2 = exact-GELU + f16 store, 3 = f32 +=
// ---------------------------------------------------------------------------
template <int EPI>
__global__ __launch_bounds__(256)
void gemm_wmma(const half_t* __restrict__ A, const half_t* __restrict__ Wt,
               const float* __restrict__ bias,
               float* __restrict__ outF, half_t* __restrict__ outH,
               int M, int N, int K, int ldOut, int colOff) {
  __shared__ __align__(16) half_t sA[64 * LDSS];   // [row][k]
  __shared__ __align__(16) half_t sB[64 * LDSS];   // [n][k]   (Wt rows)

  const int tid  = threadIdx.x;
  const int wid  = tid >> 5;
  const int lane = tid & 31;
  const int m0 = blockIdx.y * 64;
  const int n0 = blockIdx.x * 64;
  const int wm = (wid >> 1) * 16;   // wave row offset (4 waves along M)
  const int wn = (wid & 1) * 32;    // wave col offset (2 waves along N)
  const int mlane = lane & 15;
  const int hi    = lane >> 4;

  // Per-thread staging slot: row = tid/4 (64 rows), seg = tid%4 (16B each)
  const int row = tid >> 2, seg = tid & 3;
  const unsigned ldsA = (unsigned)(uintptr_t)(&sA[row * LDSS + seg * 8]);
  const unsigned ldsB = (unsigned)(uintptr_t)(&sB[row * LDSS + seg * 8]);
  const half_t* gA = A  + (size_t)(m0 + row) * K + seg * 8;
  const half_t* gB = Wt + (size_t)(n0 + row) * K + seg * 8;

  v8f acc0 = {}; v8f acc1 = {};

  for (int k0 = 0; k0 < K; k0 += 32) {
    // Async-copy 4KB A tile + 4KB B tile into LDS (one b128 per thread each)
    asm volatile("global_load_async_to_lds_b128 %0, %1, off"
                 :: "v"(ldsA), "v"((unsigned long long)(uintptr_t)(gA + k0))
                 : "memory");
    asm volatile("global_load_async_to_lds_b128 %0, %1, off"
                 :: "v"(ldsB), "v"((unsigned long long)(uintptr_t)(gB + k0))
                 : "memory");
    asm volatile("s_wait_asynccnt 0x0" ::: "memory");
    __syncthreads();

    // A fragment (16-bit A 16x32): halves k = 16*hi..16*hi+15 contiguous
    const half_t* ap = &sA[(wm + mlane) * LDSS];
    v8h alo = *(const v8h*)(ap + 16 * hi);
    v8h ahi = *(const v8h*)(ap + 16 * hi + 8);
    v16h af = __builtin_shufflevector(alo, ahi, 0, 1, 2, 3, 4, 5, 6, 7,
                                      8, 9, 10, 11, 12, 13, 14, 15);
    // B fragments (32x16): lane column n, halves k = 16*hi..16*hi+15 contiguous
    const half_t* bp0 = &sB[(wn + mlane) * LDSS + 16 * hi];
    const half_t* bp1 = &sB[(wn + 16 + mlane) * LDSS + 16 * hi];
    v8h b0lo = *(const v8h*)(bp0);
    v8h b0hi = *(const v8h*)(bp0 + 8);
    v8h b1lo = *(const v8h*)(bp1);
    v8h b1hi = *(const v8h*)(bp1 + 8);
    v16h bf0 = __builtin_shufflevector(b0lo, b0hi, 0, 1, 2, 3, 4, 5, 6, 7,
                                       8, 9, 10, 11, 12, 13, 14, 15);
    v16h bf1 = __builtin_shufflevector(b1lo, b1hi, 0, 1, 2, 3, 4, 5, 6, 7,
                                       8, 9, 10, 11, 12, 13, 14, 15);

    acc0 = __builtin_amdgcn_wmma_f32_16x16x32_f16(false, af, false, bf0, (short)0, acc0, false, false);
    acc1 = __builtin_amdgcn_wmma_f32_16x16x32_f16(false, af, false, bf1, (short)0, acc1, false, false);
    __syncthreads();
  }

  // Epilogue: acc[j] = element (m = j + 8*hi, n = mlane) of the 16x16 tile
  const int gn0 = n0 + wn + mlane;
  const int gn1 = gn0 + 16;
  const float bv0 = bias[gn0];
  const float bv1 = bias[gn1];
#pragma unroll
  for (int j = 0; j < 8; ++j) {
    int gm = m0 + wm + j + 8 * hi;
    float v0 = acc0[j] + bv0;
    float v1 = acc1[j] + bv1;
    if (EPI == 2) { // exact GELU
      v0 = 0.5f * v0 * (1.0f + erff(v0 * 0.70710678118654752f));
      v1 = 0.5f * v1 * (1.0f + erff(v1 * 0.70710678118654752f));
    }
    size_t i0 = (size_t)gm * ldOut + colOff + gn0;
    size_t i1 = (size_t)gm * ldOut + colOff + gn1;
    if (EPI == 1 || EPI == 2) { outH[i0] = (half_t)v0; outH[i1] = (half_t)v1; }
    else if (EPI == 3)        { outF[i0] += v0;        outF[i1] += v1; }
    else                      { outF[i0] = v0;         outF[i1] = v1; }
  }
}

// ---------------------------------------------------------------------------
// Generic per-row LayerNorm. mode: 0 -> f16 out; 1 -> f32 out + pos add;
// 2 -> f32 residual add (outF += LN); 3 -> f32 out
// ---------------------------------------------------------------------------
__global__ __launch_bounds__(256)
void ln_multi(const float* __restrict__ in, const float* __restrict__ g,
              const float* __restrict__ bta, half_t* __restrict__ outH,
              float* __restrict__ outF, const float* __restrict__ pos,
              int F, int mode) {
  int row = blockIdx.x, tid = threadIdx.x;
  const float* r = in + (size_t)row * F;
  float x[2] = {0.f, 0.f};
  float s = 0.f, ss = 0.f;
#pragma unroll
  for (int e = 0; e < 2; ++e) {
    int i = tid + 256 * e;
    if (i < F) { float v = r[i]; x[e] = v; s += v; ss += v * v; }
  }
  __shared__ float red[512];
  red[tid] = s; red[256 + tid] = ss;
  __syncthreads();
  for (int st = 128; st > 0; st >>= 1) {
    if (tid < st) { red[tid] += red[tid + st]; red[256 + tid] += red[256 + tid + st]; }
    __syncthreads();
  }
  float mean = red[0] / F;
  float var  = red[256] / F - mean * mean;
  float inv  = rsqrtf(var + 1e-5f);
#pragma unroll
  for (int e = 0; e < 2; ++e) {
    int i = tid + 256 * e;
    if (i < F) {
      float y = (x[e] - mean) * inv * g[i] + bta[i];
      size_t idx = (size_t)row * F + i;
      if (mode == 0)      outH[idx] = (half_t)y;
      else if (mode == 1) outF[idx] = y + pos[(size_t)(row & 255) * F + i];
      else if (mode == 2) outF[idx] += y;
      else                outF[idx] = y;
    }
  }
}

// ---------------------------------------------------------------------------
// zg[4096,1536] -> silu(z)*sigmoid(gate) -> LN(la_g, la_b) -> a_h[4096,768]
// ---------------------------------------------------------------------------
__global__ __launch_bounds__(256)
void act_ln(const float* __restrict__ zg, const float* __restrict__ lg,
            const float* __restrict__ lb, half_t* __restrict__ out) {
  int row = blockIdx.x, tid = threadIdx.x;
  const float* zr = zg + (size_t)row * 1536;
  float sv[3];
  float s = 0.f, ss = 0.f;
#pragma unroll
  for (int e = 0; e < 3; ++e) {
    int i = tid + 256 * e;                 // < 768
    float z  = zr[i];
    float gt = zr[768 + i];
    float v  = (z / (1.f + __expf(-z))) * (1.f / (1.f + __expf(-gt)));
    sv[e] = v; s += v; ss += v * v;
  }
  __shared__ float red[512];
  red[tid] = s; red[256 + tid] = ss;
  __syncthreads();
  for (int st = 128; st > 0; st >>= 1) {
    if (tid < st) { red[tid] += red[tid + st]; red[256 + tid] += red[256 + tid + st]; }
    __syncthreads();
  }
  float mean = red[0] * (1.f / 768.f);
  float var  = red[256] * (1.f / 768.f) - mean * mean;
  float inv  = rsqrtf(var + 1e-5f);
#pragma unroll
  for (int e = 0; e < 3; ++e) {
    int i = tid + 256 * e;
    out[(size_t)row * 768 + i] = (half_t)((sv[e] - mean) * inv * lg[i] + lb[i]);
  }
}

// ---------------------------------------------------------------------------
extern "C" void kernel_launch(void* const* d_in, const int* in_sizes, int n_in,
                              void* d_out, int out_size, void* d_ws, size_t ws_size,
                              hipStream_t stream) {
  const float* x       = (const float*)d_in[0];
  const float* patch_w = (const float*)d_in[1];
  const float* patch_b = (const float*)d_in[2];
  const float* pe_g    = (const float*)d_in[3];
  const float* pe_b    = (const float*)d_in[4];
  const float* pos     = (const float*)d_in[5];
  const float* n1_g    = (const float*)d_in[6];
  const float* n1_b    = (const float*)d_in[7];
  const float* pi_w    = (const float*)d_in[8];
  const float* pi_b    = (const float*)d_in[9];
  const float* po_w    = (const float*)d_in[10];
  const float* po_b    = (const float*)d_in[11];
  const float* la_g    = (const float*)d_in[12];
  const float* la_b    = (const float*)d_in[13];
  const float* mg_w    = (const float*)d_in[14];
  const float* mg_b    = (const float*)d_in[15];
  const float* sn_g    = (const float*)d_in[16];
  const float* sn_b    = (const float*)d_in[17];
  const float* n2_g    = (const float*)d_in[18];
  const float* n2_b    = (const float*)d_in[19];
  const float* m1_w    = (const float*)d_in[20];
  const float* m1_b    = (const float*)d_in[21];
  const float* m2_w    = (const float*)d_in[22];
  const float* m2_b    = (const float*)d_in[23];
  const float* fin_g   = (const float*)d_in[24];
  const float* fin_b   = (const float*)d_in[25];

  // ---- workspace layout (256B aligned slabs) ----
  char* base = (char*)d_ws;
  size_t off = 0;
  auto slab = [&](size_t bytes) { char* p = base + off; off += (bytes + 255) & ~(size_t)255; return p; };
  const size_t W_PI1 = (size_t)DD * TWODI;        //   589,824 (one direction)
  const size_t W_PO1 = (size_t)DDI * DD;          //   294,912 (one direction)
  const size_t W_PI  = 4 * W_PI1;
  const size_t W_PO  = 4 * W_PO1;
  const size_t W_MG  = (size_t)TWODI * DD;
  const size_t W_M1  = (size_t)DD * HMID;
  const size_t W_M2  = (size_t)HMID * DD;
  const size_t W_ALL = W_PI + W_PO + W_MG + W_M1 + W_M2;

  half_t* wbuf   = (half_t*)slab(W_ALL * 2);                    // per-layer f16 weights (transposed)
  float*  h      = (float*) slab((size_t)N_TOK * DD * 4);       // persistent h
  half_t* hn     = (half_t*)slab((size_t)N_TOK * DD * 2);       // hn / hn2
  float*  zg     = (float*) slab((size_t)N_TOK * TWODI * 4);    // pi output
  half_t* a_h    = (half_t*)slab((size_t)N_TOK * DDI * 2);      // act-LN out / patches
  half_t* o_h    = (half_t*)slab((size_t)N_TOK * TWODI * 2);    // concat of 4 outs
  float*  merged = (float*) slab((size_t)N_TOK * DD * 4);       // merge out / patch tmp
  half_t* mid    = (half_t*)slab((size_t)N_TOK * HMID * 2);     // MLP mid

  half_t* w_pi = wbuf;
  half_t* w_po = wbuf + W_PI;
  half_t* w_mg = wbuf + W_PI + W_PO;
  half_t* w_m1 = wbuf + W_PI + W_PO + W_MG;
  half_t* w_m2 = wbuf + W_PI + W_PO + W_MG + W_M1;

  auto nb = [](size_t n) { return dim3((unsigned)((n + 255) / 256)); };

  // ---- patch embed ----
  patchify<<<N_TOK, 256, 0, stream>>>(x, a_h);
  cvt_tr<<<nb((size_t)768 * DD), 256, 0, stream>>>(patch_w, wbuf, 768, DD);
  gemm_wmma<0><<<dim3(DD / 64, N_TOK / 64), 256, 0, stream>>>(
      a_h, wbuf, patch_b, merged, nullptr, N_TOK, DD, 768, DD, 0);
  ln_multi<<<N_TOK, 256, 0, stream>>>(merged, pe_g, pe_b, nullptr, h, pos, DD, 1);

  // ---- layers ----
  for (int l = 0; l < NLAY; ++l) {
    for (int i = 0; i < 4; ++i) {
      cvt_tr<<<nb(W_PI1), 256, 0, stream>>>(pi_w + ((size_t)(l * 4 + i)) * W_PI1,
                                            w_pi + (size_t)i * W_PI1, DD, TWODI);
      cvt_tr<<<nb(W_PO1), 256, 0, stream>>>(po_w + ((size_t)(l * 4 + i)) * W_PO1,
                                            w_po + (size_t)i * W_PO1, DDI, DD);
    }
    cvt_tr<<<nb(W_MG), 256, 0, stream>>>(mg_w + (size_t)l * W_MG, w_mg, TWODI, DD);
    cvt_tr<<<nb(W_M1), 256, 0, stream>>>(m1_w + (size_t)l * W_M1, w_m1, DD, HMID);
    cvt_tr<<<nb(W_M2), 256, 0, stream>>>(m2_w + (size_t)l * W_M2, w_m2, HMID, DD);

    // hn = LN(h, n1)
    ln_multi<<<N_TOK, 256, 0, stream>>>(h, n1_g + l * DD, n1_b + l * DD, hn, nullptr, nullptr, DD, 0);

    for (int i = 0; i < 4; ++i) {
      // zg = hn @ pi_w[l,i] + pi_b[l,i]
      gemm_wmma<0><<<dim3(TWODI / 64, N_TOK / 64), 256, 0, stream>>>(
          hn, w_pi + (size_t)i * W_PI1, pi_b + (size_t)(l * 4 + i) * TWODI,
          zg, nullptr, N_TOK, TWODI, DD, TWODI, 0);
      // a = LN(silu(z)*sigmoid(gate))
      act_ln<<<N_TOK, 256, 0, stream>>>(zg, la_g + (size_t)(l * 4 + i) * DDI,
                                        la_b + (size_t)(l * 4 + i) * DDI, a_h);
      // o_i = a @ po_w[l,i] + po_b[l,i]  -> f16 into concat buffer col i*384
      gemm_wmma<1><<<dim3(DD / 64, N_TOK / 64), 256, 0, stream>>>(
          a_h, w_po + (size_t)i * W_PO1, po_b + (size_t)(l * 4 + i) * DD,
          nullptr, o_h, N_TOK, DD, DDI, TWODI, i * DD);
    }

    // merged = concat(o) @ mg_w + mg_b
    gemm_wmma<0><<<dim3(DD / 64, N_TOK / 64), 256, 0, stream>>>(
        o_h, w_mg, mg_b + (size_t)l * DD, merged, nullptr, N_TOK, DD, TWODI, DD, 0);
    // h += LN(merged, sn)
    ln_multi<<<N_TOK, 256, 0, stream>>>(merged, sn_g + l * DD, sn_b + l * DD,
                                        nullptr, h, nullptr, DD, 2);
    // hn2 = LN(h, n2)
    ln_multi<<<N_TOK, 256, 0, stream>>>(h, n2_g + l * DD, n2_b + l * DD, hn, nullptr, nullptr, DD, 0);
    // mid = gelu(hn2 @ m1_w + m1_b)  (fused exact-GELU epilogue, f16 out)
    gemm_wmma<2><<<dim3(HMID / 64, N_TOK / 64), 256, 0, stream>>>(
        hn, w_m1, m1_b + (size_t)l * HMID, nullptr, mid, N_TOK, HMID, DD, HMID, 0);
    // h += mid @ m2_w + m2_b  (fused residual accumulate)
    gemm_wmma<3><<<dim3(DD / 64, N_TOK / 64), 256, 0, stream>>>(
        mid, w_m2, m2_b + (size_t)l * DD, h, nullptr, N_TOK, DD, HMID, DD, 0);
  }

  // ---- final LN -> d_out (f32) ----
  ln_multi<<<N_TOK, 256, 0, stream>>>(h, fin_g, fin_b, nullptr, (float*)d_out, nullptr, DD, 3);
}
